// StateDependentConv2D_27590869910068
// MI455X (gfx1250) — matrix operand order repaired and checked by reading, not compiled, not run on hardware
//
#include <hip/hip_runtime.h>
#include <hip/hip_bf16.h>

typedef __attribute__((ext_vector_type(16))) _Float16 v16h;
typedef __attribute__((ext_vector_type(8)))  float    v8f;
typedef __attribute__((ext_vector_type(4)))  int      v4i;

static constexpr int NB  = 8;    // batch
static constexpr int CH  = 16;   // in_ch == prev_ch
static constexpr int HH  = 256;
static constexpr int WW  = 256;
static constexpr int KKT = 9;    // 3x3 taps
static constexpr int HID = 64;

#if __has_builtin(__builtin_amdgcn_global_load_async_to_lds_b128)
#define USE_ASYNC_LDS 1
#else
#define USE_ASYNC_LDS 0
#endif

__device__ __forceinline__ float silu_f(float v) {
    return v * (1.0f / (1.0f + __expf(-v)));
}

// ---------------------------------------------------------------------------
// Time-embedding MLP: t[b] -> silu(Linear 1->64) -> silu(Linear 64->64)
// -> Linear 64->9. One block per batch element, 64 threads.
// ---------------------------------------------------------------------------
__global__ void temb_kernel(const float* __restrict__ t,
                            const float* __restrict__ W1, const float* __restrict__ bm1,
                            const float* __restrict__ W2, const float* __restrict__ bm2,
                            const float* __restrict__ W3, const float* __restrict__ bm3,
                            float* __restrict__ temb) {
    __shared__ float h1[HID];
    __shared__ float h2[HID];
    const int b = blockIdx.x;
    const int j = threadIdx.x;
    const float tv = t[b];
    h1[j] = silu_f(tv * W1[j] + bm1[j]);
    __syncthreads();
    float acc = bm2[j];
    #pragma unroll 8
    for (int k = 0; k < HID; ++k) acc += h1[k] * W2[k * HID + j];
    h2[j] = silu_f(acc);
    __syncthreads();
    if (j < KKT) {
        float o = bm3[j];
        #pragma unroll 8
        for (int k = 0; k < HID; ++k) o += h2[k] * W3[k * KKT + j];
        temb[b * KKT + j] = o;
    }
}

// ---------------------------------------------------------------------------
// Fused state-dependent conv.
// Block = one (b, h) row. 512 threads = 16 waves; wave w owns pixels
// [16w, 16w+16). Per wave, per non-center tap i:
//   D[16 l x 16 pix] = A_i[16 l x 16 c (f16, K padded to 32)] x
//                      prev[16 c x 16 pix]          -- v_wmma_f32_16x16x32_f16
// then k = silu(D + b1) + b2 + temb and acc += k * x[l, h+di, w+dj] (circular).
// x staging uses the CDNA5 async global->LDS path when available.
// ---------------------------------------------------------------------------
__global__ __launch_bounds__(512, 1)
void sconv_kernel(const float* __restrict__ x, const float* __restrict__ prev,
                  const float* __restrict__ A, const float* __restrict__ b1,
                  const float* __restrict__ b2, const float* __restrict__ temb,
                  float* __restrict__ out) {
    __shared__ float    xs[3][CH][WW];        // 48 KB: rows h-1, h, h+1 (circular)
    __shared__ _Float16 ps[CH][WW];           // 8 KB:  prev row, f16
    __shared__ _Float16 as[CH][KKT][CH];      // 4.5 KB: A in f16
    __shared__ float    b1s[CH * KKT];
    __shared__ float    b2s[CH * KKT];
    __shared__ float    tes[KKT];

    const int tid = threadIdx.x;
    const int b   = blockIdx.x >> 8;          // / HH
    const int h   = blockIdx.x & (HH - 1);

    // Hint the prefetch path for the next row's x tile (global_prefetch_b8).
    {
        const int hn = (h + 2) & (HH - 1);
        __builtin_prefetch(&x[(((size_t)b * CH) * HH + hn) * WW + (tid & 255)], 0, 1);
    }

    // Stage 3 circular x rows for all 16 channels (coalesced).
#if USE_ASYNC_LDS
    // CDNA5 async DMA: global -> LDS without touching VGPRs (ASYNCcnt-tracked).
    for (int q = tid; q < (3 * CH * WW) / 4; q += 512) {
        const int idx = q << 2;               // element index of a 4-float chunk
        const int r   = idx >> 12;            // / 4096
        const int c   = (idx >> 8) & 15;
        const int w   = idx & 255;
        const int hr  = (h + r + HH - 1) & (HH - 1);
        __builtin_amdgcn_global_load_async_to_lds_b128(
            (v4i*)&x[(((size_t)b * CH + c) * HH + hr) * WW + w],
            (v4i*)&xs[r][c][w],
            /*offset=*/0, /*cpol=*/0);
    }
#else
    for (int idx = tid; idx < 3 * CH * WW; idx += 512) {
        const int r  = idx >> 12;             // / 4096
        const int c  = (idx >> 8) & 15;
        const int w  = idx & 255;
        const int hr = (h + r + HH - 1) & (HH - 1);
        xs[r][c][w] = x[(((size_t)b * CH + c) * HH + hr) * WW + w];
    }
#endif
    // Stage prev row as f16 (B-matrix feed) -- needs conversion, so via VGPRs.
    for (int idx = tid; idx < CH * WW; idx += 512) {
        const int c = idx >> 8;
        const int w = idx & 255;
        ps[c][w] = (_Float16)prev[(((size_t)b * CH + c) * HH + h) * WW + w];
    }
    // Stage A (f16), biases, time embedding.
    for (int idx = tid; idx < CH * KKT * CH; idx += 512)
        ((_Float16*)as)[idx] = (_Float16)A[idx];
    for (int idx = tid; idx < CH * KKT; idx += 512) {
        b1s[idx] = b1[idx];
        b2s[idx] = b2[idx];
    }
    if (tid < KKT) tes[tid] = temb[b * KKT + tid];

#if USE_ASYNC_LDS
    // Drain this wave's async LDS writes before the block barrier publishes them.
#if __has_builtin(__builtin_amdgcn_s_wait_asynccnt)
    __builtin_amdgcn_s_wait_asynccnt(0);
#else
    asm volatile("s_wait_asynccnt 0x0" ::: "memory");
#endif
#endif
    __syncthreads();

    const int wave = tid >> 5;                // 0..15
    const int lane = tid & 31;
    const int n    = lane & 15;               // pixel within 16-wide group
    const int hi   = lane >> 4;               // half-wave select
    const int col  = (wave << 4) + n;         // absolute w of this pixel

    // B fragment: 32x16 f16, K = prev channel (upper 16 K zero-padded).
    // Layout: lanes 0-15 hold K=0..15 for column N=lane (pairs per VGPR);
    // lanes 16-31 hold K=16..31 == 0.
    v16h bfrag = {};
    if (hi == 0) {
        #pragma unroll
        for (int c = 0; c < 16; ++c) bfrag[c] = ps[c][col];
    }

    const int la   = lane & 15;               // A-matrix row M handled by this lane
    const int coff = hi ? 8 : 0;              // unused; lanes 16-31 carry zeros (K=16..31)

    float acc[8];
    #pragma unroll
    for (int j = 0; j < 8; ++j) acc[j] = 0.0f;

    const int taps[8] = {0, 1, 2, 3, 5, 6, 7, 8};   // center tap (4) is masked to 0
    #pragma unroll
    for (int ti = 0; ti < 8; ++ti) {
        const int i = taps[ti];

        // A fragment: 16x32 f16, K = prev channel (K=16..31 zero).
        v16h afrag = {};
        #pragma unroll
        for (int c = 0; c < 16; ++c) afrag[c] = hi ? (_Float16)0 : as[la][i][c];

        v8f d = {};
        d = __builtin_amdgcn_wmma_f32_16x16x32_f16(
                /*neg_a=*/false, afrag, /*neg_b=*/false, bfrag,
                /*c_mod=*/(short)0, d, /*reuse_a=*/false, /*reuse_b=*/false);

        const int   r  = i / 3;               // source row index into xs (h-1+r)
        const int   dj = (i % 3) - 1;
        const int   wc = (col + dj + WW) & (WW - 1);
        const float te = tes[i];

        // D layout: lane's VGPR j is element (M = j + 8*hi, N = lane%16).
        #pragma unroll
        for (int j = 0; j < 8; ++j) {
            const int   l   = j + (hi << 3);
            const float pre = d[j] + b1s[l * KKT + i];
            const float k   = silu_f(pre) + b2s[l * KKT + i] + te;
            acc[j] += k * xs[r][l][wc];
        }
    }

    #pragma unroll
    for (int j = 0; j < 8; ++j) {
        const int l = j + (hi << 3);
        out[(((size_t)b * CH + l) * HH + h) * WW + col] = acc[j];
    }
}

// ---------------------------------------------------------------------------
extern "C" void kernel_launch(void* const* d_in, const int* in_sizes, int n_in,
                              void* d_out, int out_size, void* d_ws, size_t ws_size,
                              hipStream_t stream) {
    const float* x    = (const float*)d_in[0];
    const float* t    = (const float*)d_in[1];
    const float* prev = (const float*)d_in[2];
    const float* A    = (const float*)d_in[3];
    const float* b1   = (const float*)d_in[4];
    const float* b2   = (const float*)d_in[5];
    const float* W1   = (const float*)d_in[6];
    const float* bm1  = (const float*)d_in[7];
    const float* W2   = (const float*)d_in[8];
    const float* bm2  = (const float*)d_in[9];
    const float* W3   = (const float*)d_in[10];
    const float* bm3  = (const float*)d_in[11];
    float* out  = (float*)d_out;
    float* temb = (float*)d_ws;               // 8*9 floats of scratch

    temb_kernel<<<NB, HID, 0, stream>>>(t, W1, bm1, W2, bm2, W3, bm3, temb);
    sconv_kernel<<<NB * HH, 512, 0, stream>>>(x, prev, A, b1, b2, temb, out);
}
